// Generator_18399639896463
// MI455X (gfx1250) — matrix operand order repaired and checked by reading, not compiled, not run on hardware
//
#include <hip/hip_runtime.h>
#include <hip/hip_bf16.h>
#include <hip/hip_fp16.h>

// ---------------------------------------------------------------------------
// 3-layer GRU (B=128,S=512,H=512,I=128) + linear head, for gfx1250 (MI455X).
//
//   * All matmuls in f16 via v_wmma_f32_16x16x32_f16 (f32 accumulate).
//   * gx = X @ W_ih^T + b_ih precomputed for the whole [B*S] batch; each wave
//     owns a 64x16 strip (B fragment reused 4x), K-loop software-pipelined so
//     next slab's loads overlap this slab's WMMAs.
//   * Recurrence: persistent kernel, 96 blocks x 8 waves == the 8x96 grid of
//     16x16 tiles of gh[128,1536]. Each block stages its 512x16 slice of
//     W_hh^T into LDS ONCE via GLOBAL_LOAD_ASYNC_TO_LDS_B128 (ASYNCcnt) and
//     reuses it for all 512 steps. Steps separated by an atomic device-scope
//     barrier (96 blocks trivially co-resident on MI455X).
// ---------------------------------------------------------------------------

typedef _Float16 f16;
typedef _Float16 half8  __attribute__((ext_vector_type(8)));
typedef _Float16 v16h   __attribute__((ext_vector_type(16)));
typedef float    v8f    __attribute__((ext_vector_type(8)));

#define B_   128
#define S_   512
#define CC_  96
#define CX_  32
#define I_   128
#define H_   512
#define G3H  1536
#define BS_  (B_ * S_)
#define REC_BLOCKS 96   // one 16-col tile of gh per block

__device__ __forceinline__ v16h cat8(half8 lo, half8 hi) {
  return __builtin_shufflevector(lo, hi, 0,1,2,3,4,5,6,7,8,9,10,11,12,13,14,15);
}

// A fragment (16x32 f16, MxK): lane l holds row m=l&15; VGPRs 0-3 hold
// K = k0 + (l>>4)*8 .. +7 (2 f16/VGPR), VGPRs 4-7 hold K = that + 16.
__device__ __forceinline__ v16h load_a_frag(const f16* __restrict__ row, int k0, int halfsel) {
  const half8* p0 = (const half8*)(row + k0 + halfsel * 8);
  const half8* p1 = (const half8*)(row + k0 + 16 + halfsel * 8);
  return cat8(*p0, *p1);
}

// B fragment (32x16 f16, KxN): lane l holds column N = l&15; elements 0-7 hold
// K = k0 + (l>>4)*8 .. +7, elements 8-15 hold K = that + 16 (p points at K row (l>>4)*8).
__device__ __forceinline__ v16h load_b_col(const f16* p, int stride) {
  v16h v;
#pragma unroll
  for (int s = 0; s < 8; ++s) { v[s] = p[s * stride]; v[s + 8] = p[(16 + s) * stride]; }
  return v;
}
__device__ __forceinline__ v16h load_b_frag(const f16* __restrict__ row_n0) {
  return load_b_col(row_n0, G3H);
}

#define WMMA_F16(a, b, c) \
  __builtin_amdgcn_wmma_f32_16x16x32_f16(false, (a), false, (b), (short)0, (c), false, false)

// Async global->LDS copy, 16B per lane. GVS mode: SGPR 64-bit base + 32-bit
// VGPR byte offset. VDST operand = LDS byte address VGPR. Tracked by ASYNCcnt.
__device__ __forceinline__ void async_copy_b128(unsigned lds_byte_off, unsigned glb_byte_off,
                                                unsigned long glb_base) {
  asm volatile("global_load_async_to_lds_b128 %0, %1, %2"
               :: "v"(lds_byte_off), "v"(glb_byte_off), "s"(glb_base) : "memory");
}
__device__ __forceinline__ void wait_async_zero() {
  asm volatile("s_wait_asynccnt 0x0" ::: "memory");
}

// ------------------------- weight transpose/convert -------------------------
// W: [G3H][K] f32 row-major  ->  Wt: [K][G3H] f16
__global__ void wt_convert_kernel(const float* __restrict__ W, f16* __restrict__ Wt, int K) {
  long i = (long)blockIdx.x * blockDim.x + threadIdx.x;
  long total = (long)K * G3H;
  if (i >= total) return;
  int k = (int)(i / G3H), n = (int)(i % G3H);
  Wt[i] = (f16)W[(long)n * K + k];
}

// ------------------------------- concat -------------------------------------
__global__ void concat_kernel(const float* __restrict__ c, const float* __restrict__ x,
                              f16* __restrict__ X) {
  long i = (long)blockIdx.x * blockDim.x + threadIdx.x;
  if (i >= (long)BS_ * I_) return;
  long row = i >> 7; int col = (int)(i & 127);
  float v = (col < CC_) ? c[row * CC_ + col] : x[row * CX_ + (col - CC_)];
  X[i] = (f16)v;
}

// --------------------------- gx = X @ W_ih^T + b ----------------------------
// grid = dim3(12, 1024), block = 256 (8 waves). Wave owns a 64x16 strip:
// n-tile = blockIdx.x*8 + wave, rows m0..m0+63. 2-stage software pipeline.
__global__ __launch_bounds__(256) void gx_gemm_kernel(
    const f16* __restrict__ X, const f16* __restrict__ Wt,
    const float* __restrict__ bias, f16* __restrict__ gx, int K) {
  const int lane  = threadIdx.x & 31;
  const int wave  = threadIdx.x >> 5;
  const int hsel  = lane >> 4;
  const int mlane = lane & 15;
  const int n0    = (blockIdx.x * 8 + wave) * 16;
  const long m0   = (long)blockIdx.y * 64;

  v8f acc[4] = {v8f{}, v8f{}, v8f{}, v8f{}};
  const f16* a0 = X + (m0 +  0 + mlane) * K;
  const f16* a1 = X + (m0 + 16 + mlane) * K;
  const f16* a2 = X + (m0 + 32 + mlane) * K;
  const f16* a3 = X + (m0 + 48 + mlane) * K;
  const f16* bbase = Wt + (long)(hsel * 8) * G3H + n0 + mlane;

  // prologue: slab k0 = 0
  v16h bf = load_b_frag(bbase);
  v16h f0 = load_a_frag(a0, 0, hsel);
  v16h f1 = load_a_frag(a1, 0, hsel);
  v16h f2 = load_a_frag(a2, 0, hsel);
  v16h f3 = load_a_frag(a3, 0, hsel);

  for (int k0 = 0; k0 < K - 32; k0 += 32) {
    const int kn = k0 + 32;
    // issue next slab's loads before consuming current fragments
    v16h bfn = load_b_frag(bbase + (long)kn * G3H);
    v16h f0n = load_a_frag(a0, kn, hsel);
    v16h f1n = load_a_frag(a1, kn, hsel);
    v16h f2n = load_a_frag(a2, kn, hsel);
    v16h f3n = load_a_frag(a3, kn, hsel);
    acc[0] = WMMA_F16(f0, bf, acc[0]);
    acc[1] = WMMA_F16(f1, bf, acc[1]);
    acc[2] = WMMA_F16(f2, bf, acc[2]);
    acc[3] = WMMA_F16(f3, bf, acc[3]);
    bf = bfn; f0 = f0n; f1 = f1n; f2 = f2n; f3 = f3n;
  }
  acc[0] = WMMA_F16(f0, bf, acc[0]);
  acc[1] = WMMA_F16(f1, bf, acc[1]);
  acc[2] = WMMA_F16(f2, bf, acc[2]);
  acc[3] = WMMA_F16(f3, bf, acc[3]);

  const float bv = bias[n0 + mlane];
  const int col = n0 + mlane;
#pragma unroll
  for (int tt = 0; tt < 4; ++tt) {
#pragma unroll
    for (int r = 0; r < 8; ++r) {
      long row = m0 + tt * 16 + r + hsel * 8;   // C/D layout: VGPR r <-> row
      gx[row * G3H + col] = (f16)(acc[tt][r] + bv);
    }
  }
}

// ------------------------------ init state ----------------------------------
__global__ void init_state_kernel(f16* __restrict__ h, unsigned* __restrict__ cnt) {
  int i = blockIdx.x * blockDim.x + threadIdx.x;
  if (i == 0) *cnt = 0u;
  if (i < B_ * H_) h[i] = (f16)0.f;
}

// --------------------------- device-wide barrier ----------------------------
__device__ __forceinline__ void grid_sync(unsigned* cnt, unsigned target) {
  __syncthreads();
  if (threadIdx.x == 0) {
    __threadfence();                     // release
    atomicAdd(cnt, 1u);
    while (atomicAdd(cnt, 0u) < target) __builtin_amdgcn_s_sleep(1);
    __threadfence();                     // acquire
  }
  __syncthreads();
}

// --------------------------- persistent recurrence --------------------------
// grid = 96 blocks x 256 threads. Block b owns gh columns [16b,16b+16).
// Wave w owns gh rows [16w,16w+16). W_hh^T slice lives in LDS for all steps.
__global__ __launch_bounds__(256) void gru_recurrent_kernel(
    const f16* __restrict__ Wt_hh,      // [H][G3H] f16
    const float* __restrict__ b_hh,     // [G3H] f32
    const f16* __restrict__ gx,         // [BS][G3H] f16 (includes b_ih)
    f16* __restrict__ h,                // [B][H] f16
    f16* __restrict__ Y,                // [B][S][H] f16 layer output
    float* __restrict__ gh,             // [B][G3H] f32 scratch
    unsigned* __restrict__ barrier_cnt) {
  __shared__ f16 ldsW[H_ * 24];         // 512 rows x 16 cols, padded to 24 f16

  const int tid   = threadIdx.x;
  const int lane  = tid & 31;
  const int wave  = tid >> 5;
  const int hsel  = lane >> 4;
  const int mlane = lane & 15;
  const int n0    = blockIdx.x * 16;
  const int m0    = wave * 16;
  const unsigned nblk = gridDim.x;

  // Stage this block's 512x16 slice of W_hh^T into LDS via async DMA
  // (16B per lane per issue; sole static LDS alloc -> LDS offset == array off).
  for (int ch = tid; ch < H_ * 2; ch += 256) {
    int k = ch >> 1, c8 = (ch & 1) * 8;
    unsigned loff = (unsigned)((k * 24 + c8) * 2);
    unsigned goff = (unsigned)(((long)k * G3H + n0 + c8) * 2);
    async_copy_b128(loff, goff, (unsigned long)Wt_hh);
  }
  wait_async_zero();
  __syncthreads();

  unsigned target = 0;
  for (int t = 0; t < S_; ++t) {
    // gh tile = h(16xK) @ Wt(Kx16), K = 512 -> 16 WMMAs, B from LDS.
    v8f acc = {};
    const f16* arow = h + (m0 + mlane) * H_;
    const f16* bp0 = ldsW + hsel * 8 * 24 + mlane;
    v16h bf = load_b_col(bp0, 24);
    v16h af = load_a_frag(arow, 0, hsel);
    for (int k0 = 0; k0 < H_ - 32; k0 += 32) {
      const f16* bp = ldsW + (k0 + 32 + hsel * 8) * 24 + mlane;
      v16h bfn = load_b_col(bp, 24);
      v16h afn = load_a_frag(arow, k0 + 32, hsel);
      acc = WMMA_F16(af, bf, acc);
      bf = bfn; af = afn;
    }
    acc = WMMA_F16(af, bf, acc);

#pragma unroll
    for (int r = 0; r < 8; ++r)
      gh[(long)(m0 + r + hsel * 8) * G3H + n0 + mlane] = acc[r];

    target += nblk;
    grid_sync(barrier_cnt, target);

    // Gate math + state update, strided over the whole grid.
    for (int idx = blockIdx.x * 256 + tid; idx < B_ * H_; idx += (int)nblk * 256) {
      int b = idx >> 9, j = idx & 511;
      const float* gr = gh + (long)b * G3H;
      float hr = gr[j]        + b_hh[j];
      float hz = gr[512 + j]  + b_hh[512 + j];
      float hn = gr[1024 + j] + b_hh[1024 + j];
      const f16* gxr = gx + ((long)b * S_ + t) * G3H;
      float xr = (float)gxr[j], xz = (float)gxr[512 + j], xn = (float)gxr[1024 + j];
      float rg = 1.f / (1.f + __expf(-(xr + hr)));
      float zg = 1.f / (1.f + __expf(-(xz + hz)));
      float ng = tanhf(xn + rg * hn);
      float hnew = (1.f - zg) * ng + zg * (float)h[idx];
      h[idx] = (f16)hnew;
      Y[((long)b * S_ + t) * H_ + j] = (f16)hnew;
    }

    target += nblk;
    grid_sync(barrier_cnt, target);
  }
}

// ------------------------------ linear head ---------------------------------
// One wave per output row; lane-strided dot over H, xor-shuffle reduce.
__global__ __launch_bounds__(256) void final_linear_kernel(
    const f16* __restrict__ Y, const float* __restrict__ Wl,
    const float* __restrict__ bl, float* __restrict__ out) {
  int lane = threadIdx.x & 31;
  int wave = threadIdx.x >> 5;
  long row = (long)blockIdx.x * 8 + wave;          // 0..BS-1
  const f16* yr = Y + row * H_;
  float s = 0.f;
  for (int j = lane; j < H_; j += 32) s += (float)yr[j] * Wl[j];
#pragma unroll
  for (int off = 16; off; off >>= 1) s += __shfl_xor(s, off, 32);
  if (lane == 0) out[row] = s + bl[0];
}

// ---------------------------------------------------------------------------
extern "C" void kernel_launch(void* const* d_in, const int* in_sizes, int n_in,
                              void* d_out, int out_size, void* d_ws, size_t ws_size,
                              hipStream_t stream) {
  (void)in_sizes; (void)n_in; (void)out_size; (void)ws_size;
  const float* c     = (const float*)d_in[0];
  const float* x     = (const float*)d_in[1];
  const float* W_ih[3] = {(const float*)d_in[2], (const float*)d_in[6], (const float*)d_in[10]};
  const float* W_hh[3] = {(const float*)d_in[3], (const float*)d_in[7], (const float*)d_in[11]};
  const float* b_ih[3] = {(const float*)d_in[4], (const float*)d_in[8], (const float*)d_in[12]};
  const float* b_hh[3] = {(const float*)d_in[5], (const float*)d_in[9], (const float*)d_in[13]};
  const float* W_lin = (const float*)d_in[14];
  const float* b_lin = (const float*)d_in[15];
  float* out = (float*)d_out;

  // ---- workspace carve-up (256B aligned) ----
  char* w = (char*)d_ws;
  auto alloc = [&](size_t bytes) { char* p = w; w += (bytes + 255) & ~(size_t)255; return p; };
  f16* Xcat = (f16*)alloc((size_t)BS_ * I_ * 2);
  f16* Ybuf[3];
  for (int l = 0; l < 3; ++l) Ybuf[l] = (f16*)alloc((size_t)BS_ * H_ * 2);
  f16*      gx  = (f16*)alloc((size_t)BS_ * G3H * 2);
  float*    gh  = (float*)alloc((size_t)B_ * G3H * 4);
  f16*      h   = (f16*)alloc((size_t)B_ * H_ * 2);
  f16* Wt_ih[3], *Wt_hh[3];
  const int Kin[3] = {I_, H_, H_};
  for (int l = 0; l < 3; ++l) {
    Wt_ih[l] = (f16*)alloc((size_t)Kin[l] * G3H * 2);
    Wt_hh[l] = (f16*)alloc((size_t)H_ * G3H * 2);
  }
  unsigned* cnt = (unsigned*)alloc(256);

  // ---- one-time prep ----
  {
    long n = (long)BS_ * I_;
    concat_kernel<<<(int)((n + 255) / 256), 256, 0, stream>>>(c, x, Xcat);
  }
  for (int l = 0; l < 3; ++l) {
    long nih = (long)Kin[l] * G3H;
    wt_convert_kernel<<<(int)((nih + 255) / 256), 256, 0, stream>>>(W_ih[l], Wt_ih[l], Kin[l]);
    long nhh = (long)H_ * G3H;
    wt_convert_kernel<<<(int)((nhh + 255) / 256), 256, 0, stream>>>(W_hh[l], Wt_hh[l], H_);
  }

  // ---- layers ----
  const f16* Xin = Xcat;
  for (int l = 0; l < 3; ++l) {
    gx_gemm_kernel<<<dim3(12, BS_ / 64), 256, 0, stream>>>(Xin, Wt_ih[l], b_ih[l], gx, Kin[l]);
    init_state_kernel<<<(B_ * H_ + 255) / 256, 256, 0, stream>>>(h, cnt);
    gru_recurrent_kernel<<<REC_BLOCKS, 256, 0, stream>>>(Wt_hh[l], b_hh[l], gx, h, Ybuf[l], gh, cnt);
    Xin = Ybuf[l];
  }

  final_linear_kernel<<<BS_ / 8, 256, 0, stream>>>(Ybuf[2], W_lin, b_lin, out);
}
